// GlobalLinearAttention_70257075028618
// MI455X (gfx1250) — matrix operand order repaired and checked by reading, not compiled, not run on hardware
//
#include <hip/hip_runtime.h>

// ---------------------------------------------------------------------------
// Types for CDNA5 WMMA (wave32, v_wmma_f32_16x16x32_bf16) and TDM descriptors
// ---------------------------------------------------------------------------
typedef __bf16 bf16;
typedef __bf16 bf16x8 __attribute__((ext_vector_type(8)));
typedef __bf16 v16bf  __attribute__((ext_vector_type(16)));
typedef float  v8f    __attribute__((ext_vector_type(8)));
typedef unsigned int u32x4 __attribute__((ext_vector_type(4)));
typedef int          i32x4 __attribute__((ext_vector_type(4)));
typedef int          i32x8 __attribute__((ext_vector_type(8)));

#define WMMA_BF16(a, b, c) \
    __builtin_amdgcn_wmma_f32_16x16x32_bf16(false, (a), false, (b), (short)0, (c), false, false)

__device__ __forceinline__ v8f vzero8() {
    return (v8f){0.f, 0.f, 0.f, 0.f, 0.f, 0.f, 0.f, 0.f};
}

// Problem constants (from reference setup_inputs)
static constexpr int BATCH  = 2;
static constexpr int SEQ    = 4096;
static constexpr int DMODEL = 1024;
static constexpr int NH     = 4;
static constexpr int DHEAD  = 256;
static constexpr int CHUNK  = 32;
static constexpr int NCH    = SEQ / CHUNK;   // 128
static constexpr int ETILE  = 64;            // e-columns of S per workgroup

// ---------------------------------------------------------------------------
// Tensor Data Mover descriptor builder (CDNA5 ISA ch.8, D# groups 0/1).
// 2D tile load Global->LDS, data_size=2B, with LDS row padding so the
// destination pitch matches our bank-conflict-free LDS layouts.
//   pad_interval_code: pad after (1<<code) * 8 bytes stored
//   pad_amount_code:   pad by (code+1) DWORDs
// ---------------------------------------------------------------------------
__device__ __forceinline__ void make_tdm_desc(u32x4& g0, i32x8& g1,
                                              unsigned lds_addr,
                                              unsigned long long gaddr,
                                              unsigned tile_d0, unsigned tile_d1,
                                              unsigned long long stride0,
                                              unsigned pad_interval_code,
                                              unsigned pad_amount_code) {
    g0[0] = 1u;                                            // count=1, user mode
    g0[1] = lds_addr;                                      // LDS byte address
    g0[2] = (unsigned)(gaddr & 0xffffffffu);               // global_addr[31:0]
    g0[3] = (unsigned)((gaddr >> 32) & 0x01ffffffu)        // global_addr[56:32]
            | (2u << 30);                                  // type=2 ("image")
    // group1 word0: data_size=2B (code 1), pad_enable, pad codes
    g1[0] = (int)((1u << 16) | (1u << 20) |
                  (pad_interval_code << 22) | (pad_amount_code << 25));
    g1[1] = (int)((tile_d0 & 0xffffu) << 16);              // tensor_dim0[15:0]
    g1[2] = (int)(((tile_d0 >> 16) & 0xffffu) |            // tensor_dim0[31:16]
                  ((tile_d1 & 0xffffu) << 16));            // tensor_dim1[15:0]
    g1[3] = (int)(((tile_d1 >> 16) & 0xffffu) |            // tensor_dim1[31:16]
                  ((tile_d0 & 0xffffu) << 16));            // tile_dim0
    g1[4] = (int)(tile_d1 & 0xffffu);                      // tile_dim1 (tile_dim2=0)
    g1[5] = (int)(unsigned)(stride0 & 0xffffffffu);        // tensor_dim0_stride[31:0]
    g1[6] = (int)(unsigned)((stride0 >> 32) & 0xffffu);    // stride[47:32] (stride1=0)
    g1[7] = 0;
}

// ---------------------------------------------------------------------------
// Fragment loaders per CDNA5 ISA 16-bit layouts (pitch in bf16 elements).
// A (16x32 MxK): lane m<16 holds K=0..7 & 16..23, lane m+16 holds K=8..15 & 24..31
// B (32x16 KxN): lane n<16 holds K=0..7 & 16..23, lane n+16 holds K=8..15 & 24..31
// ---------------------------------------------------------------------------
__device__ __forceinline__ v16bf ld_fragA(const bf16* base, int rowbase, int pitch,
                                          int koff, int lane) {
    int row = rowbase + (lane & 15);
    const bf16* p = base + row * pitch + koff + ((lane >> 4) << 3);
    bf16x8 lo = *(const bf16x8*)p;
    bf16x8 hi = *(const bf16x8*)(p + 16);
    v16bf v;
#pragma unroll
    for (int i = 0; i < 8; ++i) { v[i] = lo[i]; v[i + 8] = hi[i]; }
    return v;
}

__device__ __forceinline__ v16bf ld_fragB(const bf16* base, int rowbase, int pitch,
                                          int koff, int lane) {
    int row = rowbase + (lane & 15);
    const bf16* p = base + row * pitch + koff + ((lane >> 4) << 3);
    bf16x8 lo = *(const bf16x8*)p;
    bf16x8 hi = *(const bf16x8*)(p + 16);
    v16bf v;
#pragma unroll
    for (int i = 0; i < 8; ++i) { v[i] = lo[i]; v[i + 8] = hi[i]; }
    return v;
}

// ---------------------------------------------------------------------------
// Kernel 0: f32 -> bf16 conversion (x and weights)
// ---------------------------------------------------------------------------
__global__ __launch_bounds__(256) void cvt_f32_bf16(const float* __restrict__ in,
                                                    bf16* __restrict__ out, int n) {
    int i = (blockIdx.x * 256 + threadIdx.x) * 4;
    if (i + 3 < n) {
        float4 f = *(const float4*)(in + i);
        out[i + 0] = (bf16)f.x;
        out[i + 1] = (bf16)f.y;
        out[i + 2] = (bf16)f.z;
        out[i + 3] = (bf16)f.w;
    }
}

// ---------------------------------------------------------------------------
// Kernel 1/4: C[M,N] = op(A[M,K] @ W[N,K]^T), bf16 inputs, f32 accumulate.
// WG tile 128x128, 8 waves (4 M-waves x 2 N-waves), each wave 32x64 = 2x4 tiles.
// FMAP: relu(x)+1 epilogue. OUTBF: store bf16, else f32.
// ---------------------------------------------------------------------------
template <bool FMAP, bool OUTBF>
__global__ __launch_bounds__(256) void gemm_bf16_nt(const bf16* __restrict__ A,
                                                    const bf16* __restrict__ W,
                                                    void* __restrict__ outp,
                                                    int M, int N, int K) {
    __shared__ alignas(16) bf16 As[128 * 40];   // 80-byte pitch: conflict-free b128
    __shared__ alignas(16) bf16 Bs[128 * 40];
    const int tid = threadIdx.x, lane = tid & 31, wid = tid >> 5;
    const int wm = wid >> 1, wn = wid & 1;
    const int m0 = blockIdx.y * 128, n0 = blockIdx.x * 128;

    v8f acc[2][4];
#pragma unroll
    for (int mt = 0; mt < 2; ++mt)
#pragma unroll
        for (int nt = 0; nt < 4; ++nt) acc[mt][nt] = vzero8();

    for (int k0 = 0; k0 < K; k0 += 32) {
#pragma unroll
        for (int i = 0; i < 2; ++i) {
            int id = tid + i * 256;
            int row = id >> 2, c8 = (id & 3) << 3;
            *(bf16x8*)&As[row * 40 + c8] =
                *(const bf16x8*)&A[(size_t)(m0 + row) * K + k0 + c8];
            *(bf16x8*)&Bs[row * 40 + c8] =
                *(const bf16x8*)&W[(size_t)(n0 + row) * K + k0 + c8];
        }
        // speculative prefetch of the next k-tile (global_prefetch_b8)
        if (k0 + 32 < K) {
            __builtin_prefetch(&A[(size_t)(m0 + (tid >> 1)) * K + k0 + 32], 0, 0);
            __builtin_prefetch(&W[(size_t)(n0 + (tid >> 1)) * K + k0 + 32], 0, 0);
        }
        __syncthreads();

        v16bf af[2], bfrag[4];
#pragma unroll
        for (int mt = 0; mt < 2; ++mt)
            af[mt] = ld_fragA(As, wm * 32 + mt * 16, 40, 0, lane);
#pragma unroll
        for (int nt = 0; nt < 4; ++nt)
            bfrag[nt] = ld_fragB(Bs, wn * 64 + nt * 16, 40, 0, lane);
#pragma unroll
        for (int mt = 0; mt < 2; ++mt)
#pragma unroll
            for (int nt = 0; nt < 4; ++nt)
                acc[mt][nt] = WMMA_BF16(af[mt], bfrag[nt], acc[mt][nt]);
        __syncthreads();
    }

    const int moff = (lane >= 16) ? 8 : 0;
#pragma unroll
    for (int mt = 0; mt < 2; ++mt)
#pragma unroll
        for (int nt = 0; nt < 4; ++nt)
#pragma unroll
            for (int r = 0; r < 8; ++r) {
                int row = m0 + wm * 32 + mt * 16 + r + moff;
                int col = n0 + wn * 64 + nt * 16 + (lane & 15);
                float v = acc[mt][nt][r];
                if (FMAP) v = fmaxf(v, 0.f) + 1.f;   // relu(x)+1 feature map
                if (OUTBF) ((bf16*)outp)[(size_t)row * N + col] = (bf16)v;
                else       ((float*)outp)[(size_t)row * N + col] = v;
            }
}

// ---------------------------------------------------------------------------
// Kernel 2: per-head transpose [B*T, D](head slice) -> [B*H, dh, T] bf16
// ---------------------------------------------------------------------------
__global__ __launch_bounds__(256) void transpose_head(const bf16* __restrict__ in,
                                                      bf16* __restrict__ out) {
    __shared__ bf16 tile[32][33];
    const int bh = blockIdx.z, b = bh >> 2, h = bh & 3;
    const int tblk = blockIdx.x * 32, dblk = blockIdx.y * 32;
    for (int i = threadIdx.x; i < 1024; i += 256) {
        int r = i >> 5, c = i & 31;     // r: t, c: d
        tile[c][r] = in[(size_t)(b * SEQ + tblk + r) * DMODEL + h * DHEAD + dblk + c];
    }
    __syncthreads();
    for (int i = threadIdx.x; i < 1024; i += 256) {
        int r = i >> 5, c = i & 31;     // r: d, c: t
        out[(size_t)(bh * DHEAD + dblk + r) * SEQ + tblk + c] = tile[r][c];
    }
}

// ---------------------------------------------------------------------------
// Kernel 3: chunked causal linear attention.
// Grid: (B*H, DHEAD/ETILE) = (8,4). 256 threads = 8 waves.
// Chunk tiles (q, k, k^T, v^T) are staged Global->LDS by the Tensor Data
// Mover (one descriptor per buffer, issued by waves 0..3, TENSORcnt-tracked),
// with TDM LDS padding reproducing the bank-conflict-free pitches.
// S^T slice (64x256) lives as persistent f32 WMMA accumulators in registers
// plus a bf16 LDS shadow (B operand of out_cross = q @ S).
// ---------------------------------------------------------------------------
__global__ __launch_bounds__(256) void linattn_kernel(const bf16* __restrict__ Q,
                                                      const bf16* __restrict__ Kn,
                                                      const bf16* __restrict__ KT,
                                                      const bf16* __restrict__ VT,
                                                      bf16* __restrict__ AO) {
    constexpr int QP = DHEAD + 8;                    // 264 (528B pitch, 16B aligned)
    __shared__ alignas(16) bf16 q_s[CHUNK * QP];     // q chunk   32x256
    __shared__ alignas(16) bf16 kn_s[CHUNK * QP];    // k chunk   32x256 (row-major)
    __shared__ alignas(16) bf16 kt_s[DHEAD * 40];    // k^T chunk 256x32
    __shared__ alignas(16) bf16 vt_s[ETILE * 40];    // v^T slice 64x32
    __shared__ alignas(16) bf16 st_s[ETILE * QP];    // S^T bf16  64x256
    __shared__ alignas(16) bf16 at_s[CHUNK * 40];    // masked attn 32x32 (bf16)
    __shared__ float z_s[DHEAD];
    __shared__ float zc_s[256];
    __shared__ float den_s[CHUNK];

    const int tid = threadIdx.x, lane = tid & 31, wid = tid >> 5;
    const int bh = blockIdx.x, b = bh >> 2, h = bh & 3;
    const int e0 = blockIdx.y * ETILE;

    // out tile per wave: 2 c-tiles x 4 e-tiles
    const int omt = wid & 1, ont = wid >> 1;
    // S^T tiles per wave: e-tile (0..3) x 8 d-tiles
    const int se = wid & 3, db = (wid >> 2) * 8;

    for (int i = tid; i < ETILE * QP; i += 256) st_s[i] = (bf16)0.f;
    z_s[tid] = 0.f;
    v8f sacc[8];
#pragma unroll
    for (int j = 0; j < 8; ++j) sacc[j] = vzero8();
    __syncthreads();

    for (int nc = 0; nc < NCH; ++nc) {
        const int t0 = nc * CHUNK;

        // ---- stage chunk tiles via Tensor Data Mover (waves 0..3) ----
        if (wid < 4) {
            u32x4 g0; i32x8 g1;
            const i32x4 z4 = {0, 0, 0, 0};
            const i32x8 z8 = {0, 0, 0, 0, 0, 0, 0, 0};
            if (wid == 0) {
                // q chunk: 32 rows x 256 elems, row stride DMODEL, pitch 256->264
                make_tdm_desc(g0, g1, (unsigned)(size_t)&q_s[0],
                    (unsigned long long)(size_t)(Q + (size_t)(b * SEQ + t0) * DMODEL
                                                   + h * DHEAD),
                    DHEAD, CHUNK, (unsigned long long)DMODEL, 6u, 3u);
            } else if (wid == 1) {
                make_tdm_desc(g0, g1, (unsigned)(size_t)&kn_s[0],
                    (unsigned long long)(size_t)(Kn + (size_t)(b * SEQ + t0) * DMODEL
                                                    + h * DHEAD),
                    DHEAD, CHUNK, (unsigned long long)DMODEL, 6u, 3u);
            } else if (wid == 2) {
                // k^T chunk: 256 rows x 32 elems, row stride SEQ, pitch 32->40
                make_tdm_desc(g0, g1, (unsigned)(size_t)&kt_s[0],
                    (unsigned long long)(size_t)(KT + (size_t)(bh * DHEAD) * SEQ + t0),
                    CHUNK, DHEAD, (unsigned long long)SEQ, 3u, 3u);
            } else {
                // v^T slice: 64 rows x 32 elems
                make_tdm_desc(g0, g1, (unsigned)(size_t)&vt_s[0],
                    (unsigned long long)(size_t)(VT + (size_t)(bh * DHEAD + e0) * SEQ
                                                    + t0),
                    CHUNK, ETILE, (unsigned long long)SEQ, 3u, 3u);
            }
            __builtin_amdgcn_tensor_load_to_lds(g0, g1, z4, z4, z8, 0);
            __builtin_amdgcn_s_wait_tensorcnt(0);
        }
        __syncthreads();

        // ---- z_cross partials (uses z BEFORE this chunk's update) ----
        {
            int cq = tid & 31, seg = tid >> 5;
            float s = 0.f;
#pragma unroll
            for (int d = 0; d < 32; ++d)
                s += (float)q_s[cq * QP + seg * 32 + d] * z_s[seg * 32 + d];
            zc_s[cq * 8 + seg] = s;
        }

        // ---- attn = tril(q @ k^T), waves 0..3 (one 16x16 tile each) ----
        if (wid < 4) {
            int amt = wid & 1, ant = wid >> 1;
            v8f acc = vzero8();
#pragma unroll
            for (int ks = 0; ks < 8; ++ks) {
                v16bf a  = ld_fragA(q_s,  amt * 16, QP, ks * 32, lane);
                v16bf bb = ld_fragB(kn_s, ant * 16, QP, ks * 32, lane);
                acc = WMMA_BF16(a, bb, acc);
            }
            int moff = (lane >= 16) ? 8 : 0;
#pragma unroll
            for (int r = 0; r < 8; ++r) {
                int c  = amt * 16 + r + moff;
                int cp = ant * 16 + (lane & 15);
                at_s[c * 40 + cp] = (bf16)((cp <= c) ? acc[r] : 0.f);
            }
        }
        __syncthreads();

        // ---- denominators + z update ----
        if (tid < CHUNK) {
            float zi = 0.f;
#pragma unroll
            for (int cp = 0; cp < 32; ++cp) zi += (float)at_s[tid * 40 + cp];
            float zc = 0.f;
#pragma unroll
            for (int s2 = 0; s2 < 8; ++s2) zc += zc_s[tid * 8 + s2];
            den_s[tid] = zi + zc + 1e-6f;
        }
        {   // z[d] += sum_c k[c,d]  (rows of k^T are contiguous)
            float s = 0.f;
#pragma unroll
            for (int c = 0; c < 32; ++c) s += (float)kt_s[tid * 40 + c];
            z_s[tid] += s;
        }
        __syncthreads();

        // ---- out = (q @ S + attn @ v) / denom ----
        v8f oacc = vzero8();
#pragma unroll
        for (int ks = 0; ks < 8; ++ks) {
            v16bf a  = ld_fragA(q_s,  omt * 16, QP, ks * 32, lane);
            v16bf bb = ld_fragB(st_s, ont * 16, QP, ks * 32, lane);
            oacc = WMMA_BF16(a, bb, oacc);
        }
        {
            v16bf a  = ld_fragA(at_s, omt * 16, 40, 0, lane);
            v16bf bb = ld_fragB(vt_s, ont * 16, 40, 0, lane);
            oacc = WMMA_BF16(a, bb, oacc);
        }

        // ---- S^T += v^T @ k  (persistent register accumulators) ----
        {
            v16bf av = ld_fragA(vt_s, se * 16, 40, 0, lane);
#pragma unroll
            for (int j = 0; j < 8; ++j) {
                v16bf bk = ld_fragB(kt_s, (db + j) * 16, 40, 0, lane);
                sacc[j] = WMMA_BF16(av, bk, sacc[j]);
            }
        }

        // ---- normalize + store output (merged-head layout [B*T, D]) ----
        {
            int moff = (lane >= 16) ? 8 : 0;
#pragma unroll
            for (int r = 0; r < 8; ++r) {
                int c = omt * 16 + r + moff;
                int e = ont * 16 + (lane & 15);
                float o = oacc[r] / den_s[c];
                AO[(size_t)(b * SEQ + t0 + c) * DMODEL + h * DHEAD + e0 + e] = (bf16)o;
            }
        }
        __syncthreads();   // all reads of old st_s complete

        // ---- refresh bf16 shadow of S^T in LDS for next chunk ----
        {
            int moff = (lane >= 16) ? 8 : 0;
#pragma unroll
            for (int j = 0; j < 8; ++j)
#pragma unroll
                for (int r = 0; r < 8; ++r) {
                    int el = se * 16 + r + moff;
                    int dd = (db + j) * 16 + (lane & 15);
                    st_s[el * QP + dd] = (bf16)sacc[j][r];
                }
        }
        __syncthreads();
    }
}

// ---------------------------------------------------------------------------
// Host-side launcher
// ---------------------------------------------------------------------------
extern "C" void kernel_launch(void* const* d_in, const int* in_sizes, int n_in,
                              void* d_out, int out_size, void* d_ws, size_t ws_size,
                              hipStream_t stream) {
    const float* x  = (const float*)d_in[0];
    const float* Wq = (const float*)d_in[1];
    const float* Wk = (const float*)d_in[2];
    const float* Wv = (const float*)d_in[3];
    const float* Wo = (const float*)d_in[4];
    (void)in_sizes; (void)n_in; (void)out_size; (void)ws_size;

    const size_t BT  = (size_t)BATCH * SEQ;        // 8192
    const size_t SZX = BT * DMODEL;                // 8,388,608
    const size_t SZW = (size_t)DMODEL * DMODEL;    // 1,048,576

    char* p = (char*)d_ws;
    bf16* xb  = (bf16*)p; p += SZX * 2;
    bf16* wqb = (bf16*)p; p += SZW * 2;
    bf16* wkb = (bf16*)p; p += SZW * 2;
    bf16* wvb = (bf16*)p; p += SZW * 2;
    bf16* wob = (bf16*)p; p += SZW * 2;
    bf16* Qb  = (bf16*)p; p += SZX * 2;
    bf16* Kb  = (bf16*)p; p += SZX * 2;
    bf16* Vb  = (bf16*)p; p += SZX * 2;
    bf16* KTb = (bf16*)p; p += SZX * 2;
    bf16* VTb = (bf16*)p; p += SZX * 2;
    bf16* AOb = (bf16*)p; p += SZX * 2;

    dim3 blk(256);

    // 0) f32 -> bf16
    cvt_f32_bf16<<<(unsigned)(SZX / 1024), blk, 0, stream>>>(x,  xb,  (int)SZX);
    cvt_f32_bf16<<<(unsigned)(SZW / 1024), blk, 0, stream>>>(Wq, wqb, (int)SZW);
    cvt_f32_bf16<<<(unsigned)(SZW / 1024), blk, 0, stream>>>(Wk, wkb, (int)SZW);
    cvt_f32_bf16<<<(unsigned)(SZW / 1024), blk, 0, stream>>>(Wv, wvb, (int)SZW);
    cvt_f32_bf16<<<(unsigned)(SZW / 1024), blk, 0, stream>>>(Wo, wob, (int)SZW);

    // 1) projections: Q = relu(xWq^T)+1, K = relu(xWk^T)+1, V = xWv^T
    dim3 ggrid(DMODEL / 128, (unsigned)(BT / 128));
    gemm_bf16_nt<true,  true><<<ggrid, blk, 0, stream>>>(xb, wqb, (void*)Qb,
                                                         (int)BT, DMODEL, DMODEL);
    gemm_bf16_nt<true,  true><<<ggrid, blk, 0, stream>>>(xb, wkb, (void*)Kb,
                                                         (int)BT, DMODEL, DMODEL);
    gemm_bf16_nt<false, true><<<ggrid, blk, 0, stream>>>(xb, wvb, (void*)Vb,
                                                         (int)BT, DMODEL, DMODEL);

    // 2) per-head transposes K^T, V^T -> [B*H, dh, T]
    dim3 tgrid(SEQ / 32, DHEAD / 32, BATCH * NH);
    transpose_head<<<tgrid, blk, 0, stream>>>(Kb, KTb);
    transpose_head<<<tgrid, blk, 0, stream>>>(Vb, VTb);

    // 3) chunked causal linear attention (TDM-staged chunks)
    dim3 agrid(BATCH * NH, DHEAD / ETILE);
    linattn_kernel<<<agrid, blk, 0, stream>>>(Qb, Kb, KTb, VTb, AOb);

    // 4) output projection (f32 result)
    gemm_bf16_nt<false, false><<<ggrid, blk, 0, stream>>>(AOb, wob, d_out,
                                                          (int)BT, DMODEL, DMODEL);
}
